// MiMoV2Attention_36490042147117
// MI455X (gfx1250) — matrix operand order repaired and checked by reading, not compiled, not run on hardware
//
#include <hip/hip_runtime.h>
#include <hip/hip_bf16.h>

#define B_   2
#define S_   2048
#define HID_ 4096
#define NH_  32
#define NKV_ 8
#define HD_  128
#define VD_  96
#define DOUT_ (NH_ * VD_)   /* 3072 */

typedef _Float16 h16;
typedef __attribute__((ext_vector_type(16))) _Float16     v16h;
typedef __attribute__((ext_vector_type(8)))  float        v8f;
typedef __attribute__((ext_vector_type(4)))  unsigned int u32x4;
typedef __attribute__((ext_vector_type(4)))  float        f32x4;
typedef __attribute__((ext_vector_type(8)))  _Float16     h16x8;

union Frag { v16h v; u32x4 u[2]; };

static __device__ __forceinline__ v8f wmma_f16(const Frag& a, const Frag& b, v8f c) {
  return __builtin_amdgcn_wmma_f32_16x16x32_f16(false, a.v, false, b.v, (short)0, c, false, false);
}

// ---------------------------------------------------------------------------
// f32 -> f16 conversion (8 elements per thread iteration, b128 traffic)
// ---------------------------------------------------------------------------
__global__ void cvt_kernel(const float* __restrict__ src, h16* __restrict__ dst, long n) {
  long i = ((long)blockIdx.x * blockDim.x + threadIdx.x) * 8;
  const long stride = (long)gridDim.x * blockDim.x * 8;
  for (; i < n; i += stride) {
    f32x4 a = *(const f32x4*)(src + i);
    f32x4 c = *(const f32x4*)(src + i + 4);
    h16x8 o;
#pragma unroll
    for (int j = 0; j < 4; ++j) { o[j] = (h16)a[j]; o[j + 4] = (h16)c[j]; }
    *(h16x8*)(dst + i) = o;
  }
}

// ---------------------------------------------------------------------------
// QKV projection GEMM. X: (4096 x 4096) f16 row-major, W: (4096 x N) f16.
// One wave computes a 32(M) x 64(N) tile with 8 f32 accumulators; the B
// fragment is shared by two WMMAs (A: 0.5 b128 loads per WMMA; B gathered per element).
// Epilogue scatters to mode-specific layouts:
//   mode 0: Q  [b][h][s][d]      (N = 4096, head dim 128)
//   mode 1: Kt [b][h][d][s]      (N = 1024, transposed for QK^T B-fragments)
//   mode 2: V  [b][h][s][vd]     (N =  768, head dim 96)
// ---------------------------------------------------------------------------
__global__ __launch_bounds__(128) void qkv_gemm(const h16* __restrict__ X,
                                                const h16* __restrict__ W,
                                                h16* __restrict__ out,
                                                int N, int mode) {
  const int lane  = threadIdx.x & 31;
  const int wave  = threadIdx.x >> 5;
  const int m0    = blockIdx.x << 5;           // 32 rows per block
  const int n0    = blockIdx.y * 256 + wave * 64;
  const int mrow  = lane & 15;
  const int khalf = (lane >> 4) << 3;

  const v8f vzero = {0.f, 0.f, 0.f, 0.f, 0.f, 0.f, 0.f, 0.f};
  v8f acc[2][4];
#pragma unroll
  for (int mb = 0; mb < 2; ++mb)
#pragma unroll
    for (int t = 0; t < 4; ++t) acc[mb][t] = vzero;

  const h16* xr0 = X + (size_t)(m0 + mrow) * HID_ + khalf;
  const h16* xr1 = xr0 + (size_t)16 * HID_;
  for (int k0 = 0; k0 < HID_; k0 += 32) {
    Frag a0, a1;
    a0.u[0] = *(const u32x4*)(xr0 + k0);
    a0.u[1] = *(const u32x4*)(xr0 + k0 + 16);
    a1.u[0] = *(const u32x4*)(xr1 + k0);
    a1.u[1] = *(const u32x4*)(xr1 + k0 + 16);
    const h16* wr = W + (size_t)(k0 + khalf) * N + n0 + mrow;
    __builtin_prefetch(wr + (size_t)32 * N, 0, 1);
#pragma unroll
    for (int t = 0; t < 4; ++t) {
      Frag bf;
#pragma unroll
      for (int s = 0; s < 8; ++s) {
        bf.v[s]     = wr[(size_t)s * N + 16 * t];
        bf.v[s + 8] = wr[(size_t)(16 + s) * N + 16 * t];
      }
      acc[0][t] = wmma_f16(a0, bf, acc[0][t]);
      acc[1][t] = wmma_f16(a1, bf, acc[1][t]);
    }
  }

  const int rowoff = (lane >> 4) << 3;
#pragma unroll
  for (int mb = 0; mb < 2; ++mb) {
#pragma unroll
    for (int t = 0; t < 4; ++t) {
#pragma unroll
      for (int r = 0; r < 8; ++r) {
        const int grow = m0 + 16 * mb + r + rowoff;  // row in (B*S)
        const int gcol = n0 + 16 * t + mrow;         // col in N
        const int b = grow >> 11;                    // / S_
        const int s = grow & (S_ - 1);
        const h16 hv = (h16)acc[mb][t][r];
        if (mode == 0) {
          const int hh = gcol >> 7, d = gcol & 127;
          out[(((size_t)b * NH_ + hh) * S_ + s) * HD_ + d] = hv;
        } else if (mode == 1) {
          const int hh = gcol >> 7, d = gcol & 127;
          out[(((size_t)b * NKV_ + hh) * HD_ + d) * S_ + s] = hv;
        } else {
          const int hh = gcol / VD_, d = gcol % VD_;
          out[(((size_t)b * NKV_ + hh) * S_ + s) * VD_ + d] = hv;
        }
      }
    }
  }
}

// ---------------------------------------------------------------------------
// RoPE over first 64 head dims. mode 0: Q [b][h][s][128]; mode 1: Kt [b][h][d][s].
// One thread per rotation pair (d, d+32).
// ---------------------------------------------------------------------------
__global__ void rope_kernel(h16* __restrict__ Q, h16* __restrict__ Kt, int mode) {
  const int tid = blockIdx.x * blockDim.x + threadIdx.x;
  if (mode == 0) {
    const int d  = tid & 31;
    const int s  = (tid >> 5) & (S_ - 1);
    const int hb = tid >> 16;                    // b*NH + h  (0..63)
    const int hh = hb % NH_, b = hb / NH_;
    const size_t base = (((size_t)b * NH_ + hh) * S_ + s) * HD_;
    const float x1 = (float)Q[base + d], x2 = (float)Q[base + d + 32];
    const float ang = (float)s * __powf(1.0e6f, -(float)d * (1.0f / 32.0f));
    float sn, c; __sincosf(ang, &sn, &c);
    Q[base + d]      = (h16)(x1 * c - x2 * sn);
    Q[base + d + 32] = (h16)(x2 * c + x1 * sn);
  } else {
    const int s  = tid & (S_ - 1);
    const int d  = (tid >> 11) & 31;
    const int hb = tid >> 16;                    // b*NKV + h (0..15)
    const int hh = hb % NKV_, b = hb / NKV_;
    const size_t base = ((size_t)b * NKV_ + hh) * HD_ * S_;
    const size_t i1 = base + (size_t)d * S_ + s;
    const size_t i2 = base + (size_t)(d + 32) * S_ + s;
    const float x1 = (float)Kt[i1], x2 = (float)Kt[i2];
    const float ang = (float)s * __powf(1.0e6f, -(float)d * (1.0f / 32.0f));
    float sn, c; __sincosf(ang, &sn, &c);
    Kt[i1] = (h16)(x1 * c - x2 * sn);
    Kt[i2] = (h16)(x2 * c + x1 * sn);
  }
}

// ---------------------------------------------------------------------------
// Flash attention: one wave per 16-row q-tile, key blocks of 32.
// QK^T: 8 WMMA / block (4 d-chunks x 2 key subtiles), online softmax,
// P staged via wave-private LDS to form A-fragments, PV: 6 WMMA / block.
// ---------------------------------------------------------------------------
__global__ __launch_bounds__(256) void attn_kernel(const h16* __restrict__ Qh,
                                                   const h16* __restrict__ Kth,
                                                   const h16* __restrict__ Vh,
                                                   h16* __restrict__ Ah) {
  __shared__ __align__(16) h16 plds[8][16][32];   // per-wave 16x32 P staging
  const int lane  = threadIdx.x & 31;
  const int wave  = threadIdx.x >> 5;
  const int q0    = blockIdx.x * 128 + wave * 16;
  const int b     = blockIdx.y / NH_;
  const int hh    = blockIdx.y % NH_;
  const int hk    = hh >> 2;                      // NH/NKV = 4
  const int mrow  = lane & 15;
  const int khalf = (lane >> 4) << 3;
  const int rowoff = (lane >> 4) << 3;

  const h16* Qb = Qh + (((size_t)b * NH_ + hh) * S_ + q0) * HD_;
  const h16* Kb = Kth + ((size_t)b * NKV_ + hk) * HD_ * S_;
  const h16* Vb = Vh + ((size_t)b * NKV_ + hk) * S_ * VD_;

  // Q A-fragments for the 4 d-chunks (resident for the whole key loop)
  Frag qa[4];
  {
    const h16* qr = Qb + (size_t)mrow * HD_ + khalf;
#pragma unroll
    for (int c = 0; c < 4; ++c) {
      qa[c].u[0] = *(const u32x4*)(qr + 32 * c);
      qa[c].u[1] = *(const u32x4*)(qr + 32 * c + 16);
    }
  }

  const v8f vzero = {0.f, 0.f, 0.f, 0.f, 0.f, 0.f, 0.f, 0.f};
  v8f acc[6];
#pragma unroll
  for (int t = 0; t < 6; ++t) acc[t] = vzero;
  float rm[8], rl[8];
#pragma unroll
  for (int r = 0; r < 8; ++r) { rm[r] = -3.0e38f; rl[r] = 0.0f; }

  const float scale = 0.088388347648318447f;      // 1/sqrt(128)
  const int nkb = (q0 + 47) >> 5;                 // causal: keys 0..q0+15

  for (int kb = 0; kb < nkb; ++kb) {
    const int k0 = kb << 5;
    v8f s0 = vzero, s1 = vzero;
#pragma unroll
    for (int c = 0; c < 4; ++c) {
      const h16* kp = Kb + (size_t)(32 * c + khalf) * S_ + k0 + mrow;  // Kt row = d
      Frag b0, b1;
#pragma unroll
      for (int s = 0; s < 8; ++s) {
        b0.v[s]     = kp[(size_t)s * S_];
        b0.v[s + 8] = kp[(size_t)(16 + s) * S_];
        b1.v[s]     = kp[(size_t)s * S_ + 16];
        b1.v[s + 8] = kp[(size_t)(16 + s) * S_ + 16];
      }
      s0 = wmma_f16(qa[c], b0, s0);
      s1 = wmma_f16(qa[c], b1, s1);
    }

    // online softmax update, C-layout: row = r + rowoff, col = mrow
#pragma unroll
    for (int r = 0; r < 8; ++r) {
      const int qi = q0 + r + rowoff;
      float v0 = (float)s0[r] * scale;
      float v1 = (float)s1[r] * scale;
      if (k0 + mrow > qi)      v0 = -1.0e9f;
      if (k0 + 16 + mrow > qi) v1 = -1.0e9f;
      float mx = fmaxf(v0, v1);
#pragma unroll
      for (int off = 1; off < 16; off <<= 1)
        mx = fmaxf(mx, __shfl_xor(mx, off, 32));
      const float mnew = fmaxf(rm[r], mx);
      const float cf = __expf(rm[r] - mnew);
      const float p0 = __expf(v0 - mnew);
      const float p1 = __expf(v1 - mnew);
      float ps = p0 + p1;
#pragma unroll
      for (int off = 1; off < 16; off <<= 1)
        ps += __shfl_xor(ps, off, 32);
      rl[r] = rl[r] * cf + ps;
      rm[r] = mnew;
#pragma unroll
      for (int t = 0; t < 6; ++t) acc[t][r] *= cf;
      plds[wave][r + rowoff][mrow]      = (h16)p0;
      plds[wave][r + rowoff][mrow + 16] = (h16)p1;
    }

    // reshape P (C-layout) -> A-fragment via LDS (same-wave DS ops are in-order)
    Frag pa;
    pa.u[0] = *(const u32x4*)&plds[wave][mrow][khalf];
    pa.u[1] = *(const u32x4*)&plds[wave][mrow][khalf + 16];

    // P @ V : contraction over 32 keys, 6 N-tiles of 16 (VD = 96)
#pragma unroll
    for (int t = 0; t < 6; ++t) {
      const h16* vp = Vb + (size_t)(k0 + khalf) * VD_ + 16 * t + mrow;  // B row = key
      Frag vb;
#pragma unroll
      for (int s = 0; s < 8; ++s) {
        vb.v[s]     = vp[(size_t)s * VD_];
        vb.v[s + 8] = vp[(size_t)(16 + s) * VD_];
      }
      acc[t] = wmma_f16(pa, vb, acc[t]);
    }
  }

  // normalize and store to Ah [b][s][h*96 + vd] (f16, feeds Wo GEMM)
#pragma unroll
  for (int t = 0; t < 6; ++t) {
#pragma unroll
    for (int r = 0; r < 8; ++r) {
      const int srow = q0 + r + rowoff;
      const float o = acc[t][r] / rl[r];
      Ah[((size_t)b * S_ + srow) * DOUT_ + hh * VD_ + 16 * t + mrow] = (h16)o;
    }
  }
}

// ---------------------------------------------------------------------------
// Output projection: Ah (4096 x 3072) f16 @ Wo (3072 x 3072) f16 -> f32 out.
// One wave computes a 32(M) x 64(N) tile (8 accumulators, shared B frags).
// ---------------------------------------------------------------------------
__global__ __launch_bounds__(128) void out_gemm(const h16* __restrict__ A,
                                                const h16* __restrict__ W,
                                                float* __restrict__ out) {
  const int lane  = threadIdx.x & 31;
  const int wave  = threadIdx.x >> 5;
  const int m0    = blockIdx.x << 5;
  const int n0    = blockIdx.y * 256 + wave * 64;
  const int mrow  = lane & 15;
  const int khalf = (lane >> 4) << 3;

  const v8f vzero = {0.f, 0.f, 0.f, 0.f, 0.f, 0.f, 0.f, 0.f};
  v8f acc[2][4];
#pragma unroll
  for (int mb = 0; mb < 2; ++mb)
#pragma unroll
    for (int t = 0; t < 4; ++t) acc[mb][t] = vzero;

  const h16* ar0 = A + (size_t)(m0 + mrow) * DOUT_ + khalf;
  const h16* ar1 = ar0 + (size_t)16 * DOUT_;
  for (int k0 = 0; k0 < DOUT_; k0 += 32) {
    Frag a0, a1;
    a0.u[0] = *(const u32x4*)(ar0 + k0);
    a0.u[1] = *(const u32x4*)(ar0 + k0 + 16);
    a1.u[0] = *(const u32x4*)(ar1 + k0);
    a1.u[1] = *(const u32x4*)(ar1 + k0 + 16);
    const h16* wr = W + (size_t)(k0 + khalf) * DOUT_ + n0 + mrow;
    __builtin_prefetch(wr + (size_t)32 * DOUT_, 0, 1);
#pragma unroll
    for (int t = 0; t < 4; ++t) {
      Frag bf;
#pragma unroll
      for (int s = 0; s < 8; ++s) {
        bf.v[s]     = wr[(size_t)s * DOUT_ + 16 * t];
        bf.v[s + 8] = wr[(size_t)(16 + s) * DOUT_ + 16 * t];
      }
      acc[0][t] = wmma_f16(a0, bf, acc[0][t]);
      acc[1][t] = wmma_f16(a1, bf, acc[1][t]);
    }
  }

  const int rowoff = (lane >> 4) << 3;
#pragma unroll
  for (int mb = 0; mb < 2; ++mb) {
#pragma unroll
    for (int t = 0; t < 4; ++t) {
#pragma unroll
      for (int r = 0; r < 8; ++r) {
        out[(size_t)(m0 + 16 * mb + r + rowoff) * DOUT_ + n0 + 16 * t + mrow] =
            acc[mb][t][r];
      }
    }
  }
}

// ---------------------------------------------------------------------------
extern "C" void kernel_launch(void* const* d_in, const int* in_sizes, int n_in,
                              void* d_out, int out_size, void* d_ws, size_t ws_size,
                              hipStream_t stream) {
  (void)in_sizes; (void)n_in; (void)out_size; (void)ws_size;
  const float* X  = (const float*)d_in[0];
  /* d_in[1] attention_mask, d_in[2] position_ids: causal mask & positions are
     reconstructed analytically inside the kernels. */
  const float* Wq = (const float*)d_in[3];
  const float* Wk = (const float*)d_in[4];
  const float* Wv = (const float*)d_in[5];
  const float* Wo = (const float*)d_in[6];

  h16* ws = (h16*)d_ws;
  size_t off = 0;
  h16* Xh  = ws + off; off += (size_t)B_ * S_ * HID_;          // 16.8M
  h16* Wqh = ws + off; off += (size_t)HID_ * NH_ * HD_;        // 16.8M
  h16* Wkh = ws + off; off += (size_t)HID_ * NKV_ * HD_;       //  4.2M
  h16* Wvh = ws + off; off += (size_t)HID_ * NKV_ * VD_;       //  3.1M
  h16* Woh = ws + off; off += (size_t)DOUT_ * DOUT_;           //  9.4M
  h16* Qh  = ws + off; off += (size_t)B_ * NH_ * S_ * HD_;     // 16.8M
  h16* Kth = ws + off; off += (size_t)B_ * NKV_ * HD_ * S_;    //  4.2M
  h16* Vh  = ws + off; off += (size_t)B_ * NKV_ * S_ * VD_;    //  3.1M
  h16* Ah  = ws + off; off += (size_t)B_ * S_ * DOUT_;         // 12.6M

  // 1) f32 -> f16 operand conversion
  cvt_kernel<<<1024, 256, 0, stream>>>(X,  Xh,  (long)B_ * S_ * HID_);
  cvt_kernel<<<1024, 256, 0, stream>>>(Wq, Wqh, (long)HID_ * NH_ * HD_);
  cvt_kernel<<<512,  256, 0, stream>>>(Wk, Wkh, (long)HID_ * NKV_ * HD_);
  cvt_kernel<<<512,  256, 0, stream>>>(Wv, Wvh, (long)HID_ * NKV_ * VD_);
  cvt_kernel<<<1024, 256, 0, stream>>>(Wo, Woh, (long)DOUT_ * DOUT_);

  // 2) QKV projections (WMMA), scattering to attention layouts
  qkv_gemm<<<dim3(128, 16), 128, 0, stream>>>(Xh, Wqh, Qh,  NH_ * HD_,  0);
  qkv_gemm<<<dim3(128, 4),  128, 0, stream>>>(Xh, Wkh, Kth, NKV_ * HD_, 1);
  qkv_gemm<<<dim3(128, 3),  128, 0, stream>>>(Xh, Wvh, Vh,  NKV_ * VD_, 2);

  // 3) RoPE on Q and Kt
  rope_kernel<<<16384, 256, 0, stream>>>(Qh, Kth, 0);
  rope_kernel<<<4096,  256, 0, stream>>>(Qh, Kth, 1);

  // 4) causal flash attention (WMMA QK^T + PV)
  attn_kernel<<<dim3(S_ / 128, B_ * NH_), 256, 0, stream>>>(Qh, Kth, Vh, Ah);

  // 5) output projection (WMMA), f32 result
  out_gemm<<<dim3(256 / 2, 12), 128, 0, stream>>>(Ah, Woh, (float*)d_out);
}
